// MinkUNetBackbone_7868380086730
// MI455X (gfx1250) — matrix operand order repaired and checked by reading, not compiled, not run on hardware
//
#include <hip/hip_runtime.h>
#include <hip/hip_bf16.h>

// ---------------------------------------------------------------------------
// MinkUNet sparse-conv backbone for gfx1250 (MI455X).
//
// Gather-bandwidth-bound workload -> bf16 activations + bf16 weights, f32
// accumulation with v_wmma_f32_16x16x32_bf16 (wave32). One wave owns a
// 16-row x 64-col output strip; the gathered A fragment is reused across 4
// WMMA column tiles. The block's 64-column weight slice for the current
// kernel offset k is staged in LDS (ds_load_b128 feeds B fragments), cutting
// redundant L2 weight traffic 4x across the 4 waves of the block.
// ---------------------------------------------------------------------------

typedef __attribute__((ext_vector_type(16))) __bf16        v16bf;
typedef __attribute__((ext_vector_type(8)))  float         f32x8;
typedef __attribute__((ext_vector_type(4)))  unsigned int  u32x4;

struct Bytes32 { u32x4 a; u32x4 b; };

__device__ __forceinline__ v16bf frag_from(const __hip_bfloat16* p_lo,
                                           const __hip_bfloat16* p_hi) {
  Bytes32 t;
  t.a = *reinterpret_cast<const u32x4*>(p_lo);   // 8 bf16 = 16B (b128)
  t.b = *reinterpret_cast<const u32x4*>(p_hi);   // 8 bf16 = 16B (b128)
  return __builtin_bit_cast(v16bf, t);
}

#define TN 4          // 4 column tiles (64 cols) per wave
#define STAT_ROWS 512

extern __shared__ char smem_raw[];

// ---------------------------------------------------------------------------
// Gather-GEMM sparse conv: out[Nout,Cout] = sum_k x[nbr[:,k]] @ W[k]
//   x   : [padIdx+1, Cin] bf16 (row padIdx is all zeros)
//   wt  : [K, Cout, Cin]  bf16 (B^T layout; Cin multiple of 32)
//   nbr : [Nout, K] int32 (entry == padIdx -> zero row); nullptr = identity
// Block = 128 threads (4 waves) -> 64 rows x 64 cols.
// Dynamic LDS: 64 * (Cin + 8) bf16 (16B row pad to skew LDS banks).
// ---------------------------------------------------------------------------
__global__ __launch_bounds__(128)
void spconv_wmma(const __hip_bfloat16* __restrict__ x,
                 const __hip_bfloat16* __restrict__ wt,
                 const int* __restrict__ nbr,
                 float* __restrict__ out,
                 int Nout, int padIdx, int Cin, int Cout, int K)
{
  __hip_bfloat16* sw = (__hip_bfloat16*)smem_raw;
  const int ldsRow  = Cin + 8;                 // bf16 elements per LDS row
  const int lane    = threadIdx.x & 31;
  const int wave    = threadIdx.x >> 5;
  const int rowbase = (blockIdx.x * 4 + wave) * 16;
  const int colgrp  = blockIdx.y * (16 * TN);

  const int m     = lane & 15;                 // A row / B col within tile
  const int hsel  = lane >> 4;                 // lane half selects K sub-block
  const int abase = hsel * 8;                  // A: K offs {0..7,16..23}/{8..15,24..31}

  f32x8 acc[TN];
#pragma unroll
  for (int t = 0; t < TN; ++t) acc[t] = f32x8{0.f,0.f,0.f,0.f,0.f,0.f,0.f,0.f};

  const int  orow   = rowbase + m;
  const bool rvalid = orow < Nout;             // OOB waves gather the zero pad row

  for (int k = 0; k < K; ++k) {
    // -- cooperative weight staging: wt[k][colgrp..+64)[0..Cin) -> LDS --
    __syncthreads();                           // previous-iter LDS reads done
    const __hip_bfloat16* wk = wt + (long)k * Cout * Cin;
    for (int e = threadIdx.x * 8; e < 64 * Cin; e += 128 * 8) {
      const int col = e / Cin;                 // Cin multiple of 8 -> chunk stays in row
      const int ci  = e - col * Cin;
      u32x4 v = u32x4{0u, 0u, 0u, 0u};
      if (colgrp + col < Cout)
        v = *reinterpret_cast<const u32x4*>(wk + (long)(colgrp + col) * Cin + ci);
      *reinterpret_cast<u32x4*>(sw + col * ldsRow + ci) = v;   // ds_store_b128
    }
    __syncthreads();

    // -- gather + WMMA --
    int src = padIdx;
    if (rvalid) src = nbr ? nbr[(long)orow * K + k] : orow;
    const __hip_bfloat16* xrow = x + (long)src * Cin;

    for (int cb = 0; cb < Cin; cb += 32) {
      // A fragment: 16x32 bf16 per CDNA5 16-bit A layout
      v16bf a = frag_from(xrow + cb + abase, xrow + cb + abase + 16);
#pragma unroll
      for (int t = 0; t < TN; ++t) {
        const int col = colgrp + t * 16;
        if (col < Cout) {                      // wave-uniform
          // B fragment from LDS: col = lane&15, K offs {0..7,16..23}/{8..15,24..31}
          const __hip_bfloat16* brow = sw + (t * 16 + m) * ldsRow + cb + abase;
          v16bf b = frag_from(brow, brow + 16); // 2x ds_load_b128
          acc[t] = __builtin_amdgcn_wmma_f32_16x16x32_bf16(
              false, a, false, b, (short)0, acc[t], false, false);
        }
      }
    }
  }

  // C/D layout: lane<16 -> rows r, lane>=16 -> rows r+8; col = lane&15
  const int ccol  = lane & 15;
  const int rhalf = (lane >> 4) * 8;
#pragma unroll
  for (int t = 0; t < TN; ++t) {
    const int col = colgrp + t * 16 + ccol;
    if (col < Cout) {
#pragma unroll
      for (int r = 0; r < 8; ++r) {
        const int row = rowbase + rhalf + r;
        if (row < Nout) out[(long)row * Cout + col] = acc[t][r];
      }
    }
  }
}

// ---------------------------------------------------------------------------
// BatchNorm stats: deterministic two-stage column reduction (sum, sumsq)
// ---------------------------------------------------------------------------
__global__ __launch_bounds__(256)
void col_stats_partial(const float* __restrict__ h, int N, int C,
                       float* __restrict__ part)
{
  const int r0 = blockIdx.x * STAT_ROWS;
  const int r1 = (r0 + STAT_ROWS < N) ? r0 + STAT_ROWS : N;
  for (int c = threadIdx.x; c < C; c += 256) {
    float s = 0.f, q = 0.f;
    for (int r = r0; r < r1; ++r) {
      float v = h[(long)r * C + c];
      s += v; q += v * v;
    }
    part[(long)blockIdx.x * 2 * C + c]     = s;
    part[(long)blockIdx.x * 2 * C + C + c] = q;
  }
}

__global__ __launch_bounds__(256)
void col_stats_final(const float* __restrict__ part, int nblk, int C,
                     float* __restrict__ stats)
{
  const int c = blockIdx.x * 256 + threadIdx.x;
  if (c >= C) return;
  float s = 0.f, q = 0.f;
  for (int b = 0; b < nblk; ++b) {
    s += part[(long)b * 2 * C + c];
    q += part[(long)b * 2 * C + C + c];
  }
  stats[c] = s; stats[C + c] = q;
}

// ---------------------------------------------------------------------------
// Fused BN + residual + ReLU + bf16 quantize epilogue. Covers rows [0..N]
// (row N = zero pad row of the bf16 activation). out_ld lets the up-conv
// write straight into the left half of the concat buffer.
// ---------------------------------------------------------------------------
__global__ __launch_bounds__(256)
void bn_act(const float* __restrict__ h, const float* __restrict__ stats,
            const float* __restrict__ g, const float* __restrict__ bta,
            int N, int C, float invN,
            const float* __restrict__ res_f32,
            const __hip_bfloat16* __restrict__ res_bf, int res_ld,
            int do_relu,
            __hip_bfloat16* __restrict__ out_bf, int out_ld,
            float* __restrict__ out_f32)
{
  const long idx = (long)blockIdx.x * 256 + threadIdx.x;
  const long total = (long)(N + 1) * C;
  if (idx >= total) return;
  const int c = (int)(idx % C);
  const int r = (int)(idx / C);
  if (r == N) {                                       // zero pad row
    if (out_bf) out_bf[(long)r * out_ld + c] = __float2bfloat16(0.f);
    return;
  }
  const float mn  = stats[c] * invN;
  const float var = stats[C + c] * invN - mn * mn;
  float y = (h[idx] - mn) * rsqrtf(var + 1e-5f) * g[c] + bta[c];
  if (res_f32) y += res_f32[idx];
  if (res_bf)  y += __bfloat162float(res_bf[(long)r * res_ld + c]);
  if (do_relu) y = fmaxf(y, 0.f);
  if (out_bf)  out_bf[(long)r * out_ld + c] = __float2bfloat16(y);
  if (out_f32) out_f32[idx] = y;
}

// Convert + transpose + K-pad weights: f32 [K,Cin,Cout] -> bf16 [K,Cout,Cpad]
__global__ __launch_bounds__(256)
void prep_weight(const float* __restrict__ w, __hip_bfloat16* __restrict__ wt,
                 int K, int Cin, int Cpad, int Cout)
{
  const long idx = (long)blockIdx.x * 256 + threadIdx.x;
  const long total = (long)K * Cout * Cpad;
  if (idx >= total) return;
  const int  ci = (int)(idx % Cpad);
  const long t  = idx / Cpad;
  const int  co = (int)(t % Cout);
  const int  k  = (int)(t / Cout);
  const float v = (ci < Cin) ? w[((long)k * Cin + ci) * Cout + co] : 0.f;
  wt[idx] = __float2bfloat16(v);
}

// feats f32 [N,Cin] -> bf16 [N+1,Cpad] with channel + row zero padding
__global__ __launch_bounds__(256)
void prep_feats(const float* __restrict__ f, __hip_bfloat16* __restrict__ xb,
                int N, int Cin, int Cpad)
{
  const long idx = (long)blockIdx.x * 256 + threadIdx.x;
  const long total = (long)(N + 1) * Cpad;
  if (idx >= total) return;
  const int c = (int)(idx % Cpad);
  const int r = (int)(idx / Cpad);
  const float v = (r < N && c < Cin) ? f[(long)r * Cin + c] : 0.f;
  xb[idx] = __float2bfloat16(v);
}

// Transposed conv rulebook: nbr[i,k] = (off[i]==k) ? idx[i] : pad
__global__ __launch_bounds__(256)
void build_up_nbr(const int* __restrict__ idx8, const int* __restrict__ off8,
                  int Nf, int pad, int* __restrict__ nbr)
{
  const int i = blockIdx.x * 256 + threadIdx.x;
  if (i >= Nf) return;
  const int s = idx8[i];
  const int o = off8[i];
#pragma unroll
  for (int k = 0; k < 8; ++k) nbr[(long)i * 8 + k] = (k == o) ? s : pad;
}

// Copy lateral bf16 [N+1,C] into right half of concat buffer [N+1,dst_ld]
__global__ __launch_bounds__(256)
void concat_copy(const __hip_bfloat16* __restrict__ src, int N, int C,
                 __hip_bfloat16* __restrict__ dst, int dst_ld, int dst_off)
{
  const long idx = (long)blockIdx.x * 256 + threadIdx.x;
  const long total = (long)(N + 1) * C;
  if (idx >= total) return;
  const int c = (int)(idx % C);
  const int r = (int)(idx / C);
  dst[(long)r * dst_ld + dst_off + c] = src[idx];   // pad row of src is zero
}

// ---------------------------------------------------------------------------
// Host orchestration
// ---------------------------------------------------------------------------
struct BNp  { const float *g, *b; };
struct BlkP { const float* w1; BNp n1; const float* w2; BNp n2;
              const float* wd; BNp nd; };
struct EncP { const float* down_w; BNp down_n; BlkP b0, b1; };
struct DecP { const float* up_w;   BNp up_n;   BlkP b0, b1; };

extern "C" void kernel_launch(void* const* d_in, const int* in_sizes, int n_in,
                              void* d_out, int out_size, void* d_ws, size_t ws_size,
                              hipStream_t stream)
{
  (void)n_in; (void)out_size; (void)ws_size;
  static const int encC[5] = {32, 32, 64, 128, 256};
  static const int decC[5] = {256, 256, 128, 96, 96};
  static const int latC[4] = {128, 64, 32, 32};     // decoder-order lateral chans

  // ---- parse flattened inputs (setup_inputs() insertion order) ----
  int ip = 0;
  auto F = [&]() { return (const float*)d_in[ip++]; };
  auto I = [&]() { return (const int*)  d_in[ip++]; };
  const float* feats = F();
  const int N0 = in_sizes[0] / 4;

  auto rdBN  = [&]() { BNp n; n.g = F(); n.b = F(); return n; };
  auto rdBlk = [&](bool has_wd) {
    BlkP b; b.w1 = F(); b.n1 = rdBN(); b.w2 = F(); b.n2 = rdBN();
    if (has_wd) { b.wd = F(); b.nd = rdBN(); } else { b.wd = nullptr; b.nd = BNp{nullptr,nullptr}; }
    return b;
  };
  BlkP stem; stem.w1 = F(); stem.n1 = rdBN(); stem.w2 = F(); stem.n2 = rdBN(); stem.wd = nullptr;
  EncP enc[4];
  for (int i = 0; i < 4; ++i) {
    enc[i].down_w = F(); enc[i].down_n = rdBN();
    enc[i].b0 = rdBlk(encC[i] != encC[i + 1]);
    enc[i].b1 = rdBlk(false);
  }
  DecP dec[4];
  for (int i = 0; i < 4; ++i) {
    dec[i].up_w = F(); dec[i].up_n = rdBN();
    dec[i].b0 = rdBlk(true);
    dec[i].b1 = rdBlk(false);
  }
  const int* kmap[5]; int Nl[5];
  for (int l = 0; l < 5; ++l) { int sz = in_sizes[ip]; kmap[l] = I(); Nl[l] = sz / 27; }
  const int* downs[4]; for (int i = 0; i < 4; ++i) downs[i] = I();
  const int* upidx[4]; for (int i = 0; i < 4; ++i) upidx[i] = I();
  const int* upoff[4]; for (int i = 0; i < 4; ++i) upoff[i] = I();

  // ---- workspace bump allocator ----
  char* wsp = (char*)d_ws;
  size_t off = 0;
  auto alloc = [&](size_t bytes) -> void* {
    size_t a = (off + 255) & ~(size_t)255; off = a + bytes; return (void*)(wsp + a);
  };

  long maxH = 0, maxA = 0;
  { static const int f32w[5] = {96, 96, 128, 256, 256};   // max f32 conv-out width per level
    static const int bfw[5]  = {128, 128, 192, 384, 256}; // max bf16 act width per level
    for (int l = 0; l < 5; ++l) {
      long h = (long)Nl[l] * f32w[l];       if (h > maxH) maxH = h;
      long a = (long)(Nl[l] + 1) * bfw[l];  if (a > maxA) maxA = a;
    } }

  float* hA = (float*)alloc((size_t)maxH * 4);
  float* hB = (float*)alloc((size_t)maxH * 4);
  float* hC = (float*)alloc((size_t)maxH * 4);
  __hip_bfloat16* rot[4];
  for (int i = 0; i < 4; ++i) rot[i] = (__hip_bfloat16*)alloc((size_t)maxA * 2);
  __hip_bfloat16* latb[4];
  { static const int lw[4] = {32, 32, 64, 128};           // lateral width per level 0..3
    for (int l = 0; l < 4; ++l) latb[l] = (__hip_bfloat16*)alloc((size_t)(Nl[l] + 1) * lw[l] * 2); }
  const int nblkMax = (N0 + STAT_ROWS - 1) / STAT_ROWS;
  float* statsPart = (float*)alloc((size_t)nblkMax * 2 * 384 * 4);
  float* stats     = (float*)alloc(2 * 384 * 4);
  int*   nbr8      = (int*)alloc((size_t)N0 * 8 * 4);

  // ---- launch helpers ----
  auto prepW = [&](const float* w, int K, int Cin, int Cout) -> const __hip_bfloat16* {
    const int Cpad = (Cin + 31) & ~31;
    __hip_bfloat16* wt = (__hip_bfloat16*)alloc((size_t)K * Cout * Cpad * 2);
    const long total = (long)K * Cout * Cpad;
    prep_weight<<<(unsigned)((total + 255) / 256), 256, 0, stream>>>(w, wt, K, Cin, Cpad, Cout);
    return wt;
  };
  auto conv = [&](const __hip_bfloat16* xb, int Cin, const __hip_bfloat16* wt, int K,
                  const int* nb, int Nout, int padIdx, int Cout, float* h) {
    dim3 grid((Nout + 63) / 64, (Cout + 63) / 64);
    const size_t lds = (size_t)64 * (Cin + 8) * 2;   // staged weight slice
    spconv_wmma<<<grid, 128, lds, stream>>>(xb, wt, nb, h, Nout, padIdx, Cin, Cout, K);
  };
  auto bnact = [&](const float* h, int N, int C, BNp n,
                   const float* rf32, const __hip_bfloat16* rbf, int rld,
                   int relu, __hip_bfloat16* obf, int old_, float* of32) {
    const int nblk = (N + STAT_ROWS - 1) / STAT_ROWS;
    col_stats_partial<<<nblk, 256, 0, stream>>>(h, N, C, statsPart);
    col_stats_final<<<(C + 255) / 256, 256, 0, stream>>>(statsPart, nblk, C, stats);
    const long total = (long)(N + 1) * C;
    bn_act<<<(unsigned)((total + 255) / 256), 256, 0, stream>>>(
        h, stats, n.g, n.b, N, C, 1.f / (float)N, rf32, rbf, rld, relu, obf, old_, of32);
  };
  // residual block: xout = relu(bn(conv2(relu(bn(conv1(x))))) + idt)
  auto runBlk = [&](const BlkP& p, const __hip_bfloat16* xin, int N, int ci, int co,
                    const int* nb, int K, __hip_bfloat16* xout, float* final_f32) {
    const __hip_bfloat16* w1t = prepW(p.w1, K, ci, co);
    const __hip_bfloat16* w2t = prepW(p.w2, K, co, co);
    conv(xin, ci, w1t, K, nb, N, N, co, hA);
    bnact(hA, N, co, p.n1, nullptr, nullptr, 0, 1, rot[2], co, nullptr);   // a1
    conv(rot[2], co, w2t, K, nb, N, N, co, hB);
    if (p.wd) {
      const __hip_bfloat16* wdt = prepW(p.wd, 1, ci, co);
      conv(xin, ci, wdt, 1, nullptr, N, N, co, hA);                        // 1x1 identity path
      bnact(hA, N, co, p.nd, nullptr, nullptr, 0, 0, nullptr, 0, hC);      // idt (f32)
      bnact(hB, N, co, p.n2, hC, nullptr, 0, 1, xout, co, final_f32);
    } else {
      bnact(hB, N, co, p.n2, nullptr, xin, ci, 1, xout, co, final_f32);
    }
  };

  // ---- stem ----
  prep_feats<<<(unsigned)(((long)(N0 + 1) * 32 + 255) / 256), 256, 0, stream>>>(feats, rot[0], N0, 4, 32);
  conv(rot[0], 32, prepW(stem.w1, 27, 4, 32), 27, kmap[0], N0, N0, 32, hA);
  bnact(hA, N0, 32, stem.n1, nullptr, nullptr, 0, 1, rot[1], 32, nullptr);
  conv(rot[1], 32, prepW(stem.w2, 27, 32, 32), 27, kmap[0], N0, N0, 32, hA);
  bnact(hA, N0, 32, stem.n2, nullptr, nullptr, 0, 1, latb[0], 32, nullptr);

  // ---- encoder ----
  const __hip_bfloat16* x = latb[0];
  for (int i = 0; i < 4; ++i) {
    const int ci = encC[i], co = encC[i + 1];
    const int Nf = Nl[i], Nc = Nl[i + 1];
    conv(x, ci, prepW(enc[i].down_w, 8, ci, ci), 8, downs[i], Nc, Nf, ci, hA);
    bnact(hA, Nc, ci, enc[i].down_n, nullptr, nullptr, 0, 1, rot[0], ci, nullptr);
    runBlk(enc[i].b0, rot[0], Nc, ci, co, kmap[i + 1], 27, rot[1], nullptr);
    __hip_bfloat16* outb = (i < 3) ? latb[i + 1] : rot[0];
    runBlk(enc[i].b1, rot[1], Nc, co, co, kmap[i + 1], 27, outb, nullptr);
    x = outb;
  }

  // ---- decoder ----
  for (int i = 0; i < 4; ++i) {
    const int f = 3 - i;
    const int ciU = decC[i], coU = decC[i + 1], lc = latC[i];
    const int Nf = Nl[f], Nc = Nl[f + 1];
    build_up_nbr<<<(Nf + 255) / 256, 256, 0, stream>>>(upidx[i], upoff[i], Nf, Nc, nbr8);
    conv(x, ciU, prepW(dec[i].up_w, 8, ciU, coU), 8, nbr8, Nf, Nc, coU, hA);
    const int Ctot = coU + lc;
    bnact(hA, Nf, coU, dec[i].up_n, nullptr, nullptr, 0, 1, rot[3], Ctot, nullptr);
    concat_copy<<<(unsigned)(((long)(Nf + 1) * lc + 255) / 256), 256, 0, stream>>>(
        latb[f], Nf, lc, rot[3], Ctot, coU);
    const int slot = (x == rot[0]) ? 0 : 1;
    runBlk(dec[i].b0, rot[3], Nf, Ctot, coU, kmap[f], 27, rot[slot], nullptr);
    const bool last = (i == 3);
    runBlk(dec[i].b1, rot[slot], Nf, coU, coU, kmap[f], 27, rot[1 - slot],
           last ? (float*)d_out : nullptr);
    x = rot[1 - slot];
  }
}
